// PAConv_46540265620115
// MI455X (gfx1250) — matrix operand order repaired and checked by reading, not compiled, not run on hardware
//
#include <hip/hip_runtime.h>

#define KNN_K 20
#define NPTS  1024
#define NB    4
#define NPTOT (NB * NPTS)   // 4096 rows

typedef _Float16 v16h __attribute__((ext_vector_type(16)));
typedef _Float16 v8h  __attribute__((ext_vector_type(8)));
typedef float    v8f  __attribute__((ext_vector_type(8)));

// ---------------------------------------------------------------------------
// 1) transpose x [B,3,N] -> pts [B*N,3], squared norms
// ---------------------------------------------------------------------------
__global__ void __launch_bounds__(128)
prep_kernel(const float* __restrict__ x, float* __restrict__ pts, float* __restrict__ sq)
{
    int p = blockIdx.x * blockDim.x + threadIdx.x;
    if (p >= NPTOT) return;
    int b = p >> 10, n = p & 1023;
    float vx = x[b * 3072 + 0    + n];
    float vy = x[b * 3072 + 1024 + n];
    float vz = x[b * 3072 + 2048 + n];
    pts[p * 3 + 0] = vx; pts[p * 3 + 1] = vy; pts[p * 3 + 2] = vz;
    sq[p] = vx * vx + vy * vy + vz * vz;
}

// ---------------------------------------------------------------------------
// 2) KNN: per point, insertion sort of 20 nearest (incl. self)
// ---------------------------------------------------------------------------
__global__ void __launch_bounds__(128)
knn_kernel(const float* __restrict__ pts, const float* __restrict__ sq, int* __restrict__ idx)
{
    int p = blockIdx.x * blockDim.x + threadIdx.x;
    if (p >= NPTOT) return;
    int b = p >> 10, n = p & 1023;
    const float* P  = pts + b * NPTS * 3;
    const float* SQ = sq + b * NPTS;
    float cx = P[n * 3], cy = P[n * 3 + 1], cz = P[n * 3 + 2];
    float sn = SQ[n];
    float d[KNN_K]; int id[KNN_K];
#pragma unroll
    for (int k = 0; k < KNN_K; ++k) { d[k] = 1e30f; id[k] = 0; }
    for (int m = 0; m < NPTS; ++m) {
        float dot = cx * P[m * 3] + cy * P[m * 3 + 1] + cz * P[m * 3 + 2];
        float d2 = sn - 2.0f * dot + SQ[m];
        if (d2 < d[KNN_K - 1]) {
            d[KNN_K - 1] = d2; id[KNN_K - 1] = m;
#pragma unroll
            for (int k = KNN_K - 1; k > 0; --k) {
                if (d[k] < d[k - 1]) {
                    float td = d[k]; d[k] = d[k - 1]; d[k - 1] = td;
                    int   ti = id[k]; id[k] = id[k - 1]; id[k - 1] = ti;
                }
            }
        }
    }
#pragma unroll
    for (int k = 0; k < KNN_K; ++k) idx[p * KNN_K + k] = id[k];
}

// ---------------------------------------------------------------------------
// 3) scorenet geometry features (xyz[...,10]) + conv1(6->64)+BN+ReLU+max_k
//    writes x1 into xx columns [0,64)
// ---------------------------------------------------------------------------
__global__ void __launch_bounds__(128)
feat_conv1_kernel(const float* __restrict__ pts, const int* __restrict__ idx,
                  const float* __restrict__ w,  const float* __restrict__ bconv,
                  const float* __restrict__ g,  const float* __restrict__ bb,
                  float* __restrict__ xyz, _Float16* __restrict__ xx)
{
    __shared__ float sW[64 * 6], sB[64], sG[64], sBb[64];
    for (int i = threadIdx.x; i < 384; i += blockDim.x) sW[i] = w[i];
    for (int i = threadIdx.x; i < 64; i += blockDim.x) {
        sB[i] = bconv[i];
        sG[i] = g[i] * rsqrtf(1.0f + 1e-5f);
        sBb[i] = bb[i];
    }
    __syncthreads();
    int p = blockIdx.x * blockDim.x + threadIdx.x;
    if (p >= NPTOT) return;
    int b = p >> 10, n = p & 1023;
    const float* P = pts + b * NPTS * 3;
    float cx = P[n * 3], cy = P[n * 3 + 1], cz = P[n * 3 + 2];
    float best[64];
#pragma unroll
    for (int c = 0; c < 64; ++c) best[c] = 0.0f;   // post-ReLU values are >= 0
    for (int k = 0; k < KNN_K; ++k) {
        int j = idx[p * KNN_K + k];
        float nx = P[j * 3], ny = P[j * 3 + 1], nz = P[j * 3 + 2];
        float rx = nx - cx, ry = ny - cy, rz = nz - cz;
        float dist = sqrtf(rx * rx + ry * ry + rz * rz);
        float* X = xyz + (p * KNN_K + k) * 10;
        X[0] = cx; X[1] = cy; X[2] = cz;
        X[3] = nx; X[4] = ny; X[5] = nz;
        X[6] = rx; X[7] = ry; X[8] = rz; X[9] = dist;
        float f[6] = { rx, ry, rz, cx, cy, cz };
        for (int c = 0; c < 64; ++c) {
            float h = sB[c];
#pragma unroll
            for (int q = 0; q < 6; ++q) h += f[q] * sW[c * 6 + q];
            h = h * sG[c] + sBb[c];
            h = h > 0.0f ? h : 0.0f;
            best[c] = h > best[c] ? h : best[c];
        }
    }
    _Float16* O = xx + p * 320;
#pragma unroll
    for (int c = 0; c < 64; ++c) O[c] = (_Float16)best[c];
}

// ---------------------------------------------------------------------------
// 4a) generic f32 -> f16 convert (dense weights; already [Nout,K] row-major)
// ---------------------------------------------------------------------------
__global__ void __launch_bounds__(256)
f32_to_f16_kernel(const float* __restrict__ in, _Float16* __restrict__ out, int n)
{
    int t = blockIdx.x * blockDim.x + threadIdx.x;
    if (t < n) out[t] = (_Float16)in[t];
}

// 4b) stage weight: W[o,k] for o<512 = mat[k,o]+mat[k+64,o] (pm via rep=concat(x,x));
//                   for o>=512    = mat[k,o-512]            (cm)
__global__ void __launch_bounds__(256)
build_stage_w_kernel(const float* __restrict__ mat, _Float16* __restrict__ Wst)
{
    int t = blockIdx.x * blockDim.x + threadIdx.x;
    if (t >= 1024 * 64) return;
    int k = t & 63;
    int o = t >> 6;
    float v;
    if (o < 512) v = mat[k * 512 + o] + mat[(k + 64) * 512 + o];
    else         v = mat[k * 512 + (o - 512)];
    Wst[o * 64 + k] = (_Float16)v;
}

// ---------------------------------------------------------------------------
// 5) scorenet: xyz[...,10] -> 16 (BN,ReLU) -> 8 (+bias) -> softmax over 8
// ---------------------------------------------------------------------------
__global__ void __launch_bounds__(256)
scorenet_kernel(const float* __restrict__ xyz,
                const float* __restrict__ w1, const float* __restrict__ bng,
                const float* __restrict__ bnb,
                const float* __restrict__ w2, const float* __restrict__ b2,
                float* __restrict__ sc)
{
    __shared__ float sw1[160], sg[16], sb[16], sw2[128], sb2[8];
    for (int i = threadIdx.x; i < 160; i += blockDim.x) sw1[i] = w1[i];
    for (int i = threadIdx.x; i < 128; i += blockDim.x) sw2[i] = w2[i];
    for (int i = threadIdx.x; i < 16; i += blockDim.x) {
        sg[i] = bng[i] * rsqrtf(1.0f + 1e-5f);
        sb[i] = bnb[i];
    }
    for (int i = threadIdx.x; i < 8; i += blockDim.x) sb2[i] = b2[i];
    __syncthreads();
    int t = blockIdx.x * blockDim.x + threadIdx.x;   // 0 .. 4096*20-1
    if (t >= NPTOT * KNN_K) return;
    const float* X = xyz + t * 10;
    float xv[10];
#pragma unroll
    for (int i = 0; i < 10; ++i) xv[i] = X[i];
    float h[16];
#pragma unroll
    for (int j = 0; j < 16; ++j) {
        float a = 0.0f;
#pragma unroll
        for (int i = 0; i < 10; ++i) a += xv[i] * sw1[j * 10 + i];
        a = a * sg[j] + sb[j];
        h[j] = a > 0.0f ? a : 0.0f;
    }
    float s[8];
    float mx = -1e30f;
#pragma unroll
    for (int m = 0; m < 8; ++m) {
        float a = sb2[m];
#pragma unroll
        for (int j = 0; j < 16; ++j) a += h[j] * sw2[m * 16 + j];
        s[m] = a;
        mx = a > mx ? a : mx;
    }
    float sum = 0.0f;
#pragma unroll
    for (int m = 0; m < 8; ++m) { s[m] = __expf(s[m] - mx); sum += s[m]; }
    float inv = 1.0f / sum;
    float* O = sc + t * 8;
#pragma unroll
    for (int m = 0; m < 8; ++m) O[m] = s[m] * inv;
}

// ---------------------------------------------------------------------------
// 6) WMMA GEMM: out[r,o] = sum_k A[r,k] * W[o,k]   (A,W f16; acc f32)
//    One wave computes a 16x64 tile (4 N-subtiles): each A fragment is reused
//    by 4 independent v_wmma_f32_16x16x32_f16 ops. Out-of-range columns are
//    pointer-clamped (safe loads) and simply not stored — no in-loop masking.
//    Epilogue: +bias, BN (g*rsqrt(1+eps), b), ReLU; f16 row-major and/or
//    f32 transposed to [B, Nout, 1024].
// ---------------------------------------------------------------------------
__global__ void __launch_bounds__(128)
gemm_wmma_kernel(const _Float16* __restrict__ A, int lda,
                 const _Float16* __restrict__ W, int Kdim, int Nout, int Mrows,
                 const float* __restrict__ bias,
                 const float* __restrict__ bng, const float* __restrict__ bnb,
                 int relu,
                 _Float16* __restrict__ outH, int ldo,
                 float* __restrict__ outT)
{
    const int lane = threadIdx.x & 31;
    const int wave = threadIdx.x >> 5;
    const int groupsN = (Nout + 63) >> 6;
    const int tilesM  = Mrows >> 4;
    const int tile = blockIdx.x * 4 + wave;
    if (tile >= tilesM * groupsN) return;
    const int tm = tile / groupsN;
    const int tg = tile % groupsN;

    const int half = lane >> 4;   // 0: first K-half group, 1: second
    const int l16  = lane & 15;

    const _Float16* aptr = A + (tm * 16 + l16) * lda;

    int cols[4];
    const _Float16* bptr[4];
#pragma unroll
    for (int j = 0; j < 4; ++j) {
        int c = tg * 64 + j * 16 + l16;
        cols[j] = c;
        bptr[j] = W + (c < Nout ? c : 0) * Kdim;   // clamp: safe loads, store guarded
    }

    v8f acc0 = {}, acc1 = {}, acc2 = {}, acc3 = {};
    for (int k0 = 0; k0 < Kdim; k0 += 32) {
        // A fragment: lanes 0-15 row M=l16 K {0..7,16..23}; lanes 16-31 K {8..15,24..31}
        v8h alo = *(const v8h*)(aptr + k0 + half * 8);
        v8h ahi = *(const v8h*)(aptr + k0 + 16 + half * 8);
        v16h av;
#pragma unroll
        for (int i = 0; i < 8; ++i) { av[i] = alo[i]; av[8 + i] = ahi[i]; }

        // B fragment: lanes 0-15 col N K {0..7,16..23}; lanes 16-31 K {8..15,24..31}
#define BSTEP(J, ACC)                                                         \
        {                                                                     \
            v8h b0 = *(const v8h*)(bptr[J] + k0 + half * 8);                  \
            v8h b1 = *(const v8h*)(bptr[J] + k0 + 16 + half * 8);             \
            v16h bv;                                                          \
            _Pragma("unroll")                                                 \
            for (int i = 0; i < 8; ++i) { bv[i] = b0[i]; bv[8 + i] = b1[i]; } \
            ACC = __builtin_amdgcn_wmma_f32_16x16x32_f16(                     \
                false, av, false, bv, (short)0, ACC, false, false);           \
        }
        BSTEP(0, acc0)
        BSTEP(1, acc1)
        BSTEP(2, acc2)
        BSTEP(3, acc3)
#undef BSTEP
    }

    v8f accs[4] = { acc0, acc1, acc2, acc3 };
#pragma unroll
    for (int j = 0; j < 4; ++j) {
        const int col = cols[j];
        if (col >= Nout) continue;
        float bsv = bias ? bias[col] : 0.0f;
        float gs = 1.0f, gb = 0.0f;
        if (bng) { gs = bng[col] * rsqrtf(1.0f + 1e-5f); gb = bnb[col]; }
#pragma unroll
        for (int r = 0; r < 8; ++r) {
            int row = tm * 16 + r + half * 8;
            float v = accs[j][r] + bsv;
            if (bng) v = v * gs + gb;
            if (relu) v = v > 0.0f ? v : 0.0f;
            if (outH) outH[row * ldo + col] = (_Float16)v;
            if (outT) {
                int b = row >> 10, n = row & 1023;
                outT[(b * Nout + col) * 1024 + n] = v;
            }
        }
    }
}

// ---------------------------------------------------------------------------
// 7) PAConv assemble: out[p,c] = sum_m [ sum_k sc*pm[nbr] + (sum_k sc)*cm ]
//    pm = pmcm cols [0,512), cm = cols [512,1024); then BN+ReLU -> xx col block
// ---------------------------------------------------------------------------
__global__ void __launch_bounds__(256)
assemble_kernel(const _Float16* __restrict__ pmcm, const float* __restrict__ sc,
                const int* __restrict__ idx,
                const float* __restrict__ bng, const float* __restrict__ bnb,
                _Float16* __restrict__ xx, int colOff)
{
    int t = blockIdx.x * blockDim.x + threadIdx.x;
    if (t >= NPTOT * 64) return;
    int c = t & 63;
    int p = t >> 6;
    int b = p >> 10;
    const float* S = sc + p * KNN_K * 8;
    const int*   I = idx + p * KNN_K;
    float out = 0.0f;
#pragma unroll
    for (int m = 0; m < 8; ++m) {
        float accm = 0.0f, ss = 0.0f;
        for (int k = 0; k < KNN_K; ++k) {
            float s = S[k * 8 + m];
            int j = I[k];
            accm += s * (float)pmcm[(b * NPTS + j) * 1024 + m * 64 + c];
            ss += s;
        }
        out += accm + ss * (float)pmcm[p * 1024 + 512 + m * 64 + c];
    }
    float v = out * (bng[c] * rsqrtf(1.0f + 1e-5f)) + bnb[c];
    v = v > 0.0f ? v : 0.0f;
    xx[p * 320 + colOff + c] = (_Float16)v;
}

// ---------------------------------------------------------------------------
// host launcher
// ---------------------------------------------------------------------------
static inline int cdiv(int a, int b) { return (a + b - 1) / b; }

extern "C" void kernel_launch(void* const* d_in, const int* in_sizes, int n_in,
                              void* d_out, int out_size, void* d_ws, size_t ws_size,
                              hipStream_t stream)
{
    (void)in_sizes; (void)n_in; (void)out_size; (void)ws_size;

    const float* x        = (const float*)d_in[0];
    const float* conv1_w  = (const float*)d_in[1];
    const float* conv1_b  = (const float*)d_in[2];
    const float* bn_c1_g  = (const float*)d_in[3];
    const float* bn_c1_b  = (const float*)d_in[4];
    // stages s=2..5 at base 5 + i*8: mat, sn_w1, sn_bng, sn_bnb, sn_w2, sn_b2, bn_g, bn_b
    const float* convt_w  = (const float*)d_in[37];
    const float* bnt_g    = (const float*)d_in[38];
    const float* bnt_b    = (const float*)d_in[39];
    const float* h1_w     = (const float*)d_in[40];
    const float* h1_b     = (const float*)d_in[41];
    const float* h1_g     = (const float*)d_in[42];
    const float* h1_bb    = (const float*)d_in[43];
    const float* h2_w     = (const float*)d_in[44];
    const float* h2_b     = (const float*)d_in[45];
    const float* h2_g     = (const float*)d_in[46];
    const float* h2_bb    = (const float*)d_in[47];
    const float* h3_w     = (const float*)d_in[48];
    const float* h3_b     = (const float*)d_in[49];

    // workspace layout
    char* base = (char*)d_ws;
    size_t off = 0;
    auto take = [&](size_t bytes) -> void* {
        void* p = base + off;
        off += (bytes + 255) & ~(size_t)255;
        return p;
    };
    float*     pts    = (float*)take(NPTOT * 3 * sizeof(float));
    float*     sq     = (float*)take(NPTOT * sizeof(float));
    int*       idx    = (int*)take(NPTOT * KNN_K * sizeof(int));
    float*     xyz    = (float*)take((size_t)NPTOT * KNN_K * 10 * sizeof(float));
    float*     sc     = (float*)take((size_t)NPTOT * KNN_K * 8 * sizeof(float));
    _Float16*  xx     = (_Float16*)take((size_t)NPTOT * 320 * sizeof(_Float16));
    _Float16*  pmcm   = (_Float16*)take((size_t)NPTOT * 1024 * sizeof(_Float16));
    _Float16*  xc     = (_Float16*)take((size_t)NPTOT * 1024 * sizeof(_Float16));
    _Float16*  hb1    = (_Float16*)take((size_t)NPTOT * 512 * sizeof(_Float16));
    _Float16*  hb2    = (_Float16*)take((size_t)NPTOT * 256 * sizeof(_Float16));
    _Float16*  Wst[4];
    for (int i = 0; i < 4; ++i) Wst[i] = (_Float16*)take(1024 * 64 * sizeof(_Float16));
    _Float16*  convtH = (_Float16*)take(1024 * 320 * sizeof(_Float16));
    _Float16*  h1H    = (_Float16*)take(512 * 1024 * sizeof(_Float16));
    _Float16*  h2H    = (_Float16*)take(256 * 512 * sizeof(_Float16));
    _Float16*  h3H    = (_Float16*)take(68 * 256 * sizeof(_Float16));

    // graph build + x1
    prep_kernel<<<cdiv(NPTOT, 128), 128, 0, stream>>>(x, pts, sq);
    knn_kernel<<<cdiv(NPTOT, 128), 128, 0, stream>>>(pts, sq, idx);
    feat_conv1_kernel<<<cdiv(NPTOT, 128), 128, 0, stream>>>(
        pts, idx, conv1_w, conv1_b, bn_c1_g, bn_c1_b, xyz, xx);

    // weight prep
    for (int i = 0; i < 4; ++i) {
        const float* mat = (const float*)d_in[5 + i * 8];
        build_stage_w_kernel<<<cdiv(1024 * 64, 256), 256, 0, stream>>>(mat, Wst[i]);
    }
    f32_to_f16_kernel<<<cdiv(1024 * 320, 256), 256, 0, stream>>>(convt_w, convtH, 1024 * 320);
    f32_to_f16_kernel<<<cdiv(512 * 1024, 256), 256, 0, stream>>>(h1_w, h1H, 512 * 1024);
    f32_to_f16_kernel<<<cdiv(256 * 512, 256), 256, 0, stream>>>(h2_w, h2H, 256 * 512);
    f32_to_f16_kernel<<<cdiv(68 * 256, 256), 256, 0, stream>>>(h3_w, h3H, 68 * 256);

    // PAConv stages
    for (int i = 0; i < 4; ++i) {
        const int bidx = 5 + i * 8;
        const float* sn_w1  = (const float*)d_in[bidx + 1];
        const float* sn_bng = (const float*)d_in[bidx + 2];
        const float* sn_bnb = (const float*)d_in[bidx + 3];
        const float* sn_w2  = (const float*)d_in[bidx + 4];
        const float* sn_b2  = (const float*)d_in[bidx + 5];
        const float* bn_g   = (const float*)d_in[bidx + 6];
        const float* bn_b   = (const float*)d_in[bidx + 7];

        scorenet_kernel<<<cdiv(NPTOT * KNN_K, 256), 256, 0, stream>>>(
            xyz, sn_w1, sn_bng, sn_bnb, sn_w2, sn_b2, sc);

        // pm|cm = x_in[4096,64] x Wst[64,1024]   (A cols 64*i in xx slab)
        {
            int tiles = (NPTOT / 16) * cdiv(1024, 64);
            gemm_wmma_kernel<<<cdiv(tiles, 4), 128, 0, stream>>>(
                xx + 64 * i, 320, Wst[i], 64, 1024, NPTOT,
                nullptr, nullptr, nullptr, 0, pmcm, 1024, nullptr);
        }
        assemble_kernel<<<cdiv(NPTOT * 64, 256), 256, 0, stream>>>(
            pmcm, sc, idx, bn_g, bn_b, xx, 64 * (i + 1));
    }

    // convt: [4096,320] x [320,1024] -> xc (BN+ReLU)
    {
        int tiles = (NPTOT / 16) * cdiv(1024, 64);
        gemm_wmma_kernel<<<cdiv(tiles, 4), 128, 0, stream>>>(
            xx, 320, convtH, 320, 1024, NPTOT,
            nullptr, bnt_g, bnt_b, 1, xc, 1024, nullptr);
    }
    // h1: [4096,1024] x [1024,512] -> hb1 (+bias, BN, ReLU)
    {
        int tiles = (NPTOT / 16) * cdiv(512, 64);
        gemm_wmma_kernel<<<cdiv(tiles, 4), 128, 0, stream>>>(
            xc, 1024, h1H, 1024, 512, NPTOT,
            h1_b, h1_g, h1_bb, 1, hb1, 512, nullptr);
    }
    // h2: [4096,512] x [512,256] -> hb2 (+bias, BN, ReLU)
    {
        int tiles = (NPTOT / 16) * cdiv(256, 64);
        gemm_wmma_kernel<<<cdiv(tiles, 4), 128, 0, stream>>>(
            hb1, 512, h2H, 512, 256, NPTOT,
            h2_b, h2_g, h2_bb, 1, hb2, 256, nullptr);
    }
    // h3: [4096,256] x [256,68] -> d_out transposed [B,68,N] (f32, +bias)
    {
        int tiles = (NPTOT / 16) * cdiv(68, 64);
        gemm_wmma_kernel<<<cdiv(tiles, 4), 128, 0, stream>>>(
            hb2, 256, h3H, 256, 68, NPTOT,
            h3_b, nullptr, nullptr, 0, nullptr, 0, (float*)d_out);
    }
}
